// GroupedQueryAttention_40175124087437
// MI455X (gfx1250) — matrix operand order repaired and checked by reading, not compiled, not run on hardware
//
#include <hip/hip_runtime.h>

// ---------------------------------------------------------------------------
// GQA forward for MI455X (gfx1250, wave32, WMMA).
// All heavy math in v_wmma_f32_16x16x32_bf16 (f32 accumulation).
// Tile staging uses GLOBAL_LOAD_ASYNC_TO_LDS (ASYNCcnt) when the toolchain
// exposes the builtin; otherwise falls back to load_b128 + ds_store_b128.
// ---------------------------------------------------------------------------

typedef __bf16 bf16;
typedef __attribute__((ext_vector_type(16))) __bf16 bf16x16;
typedef __attribute__((ext_vector_type(8)))  float   f32x8;
typedef __attribute__((ext_vector_type(4)))  float   f32x4;
typedef __attribute__((ext_vector_type(4)))  int     i32x4;

union FragU { bf16x16 v; f32x4 f[2]; };

constexpr int BATCH = 2;
constexpr int SEQ   = 2048;
constexpr int CDIM  = 2048;
constexpr int NH    = 32;
constexpr int NKV   = 4;
constexpr int HD    = 64;
constexpr int NREP  = NH / NKV;

#if defined(__has_builtin)
#if __has_builtin(__builtin_amdgcn_global_load_async_to_lds_b128) && \
    __has_builtin(__builtin_amdgcn_s_wait_asynccnt)
#define HAVE_ASYNC_LDS 1
#endif
#endif

#define AS1 __attribute__((address_space(1)))
#define AS3 __attribute__((address_space(3)))

__device__ __forceinline__ f32x8 wmma_bf16(bf16x16 a, bf16x16 b, f32x8 c) {
  // D = A(16x32) * B(32x16) + C(16x16), f32 accumulate
  return __builtin_amdgcn_wmma_f32_16x16x32_bf16(false, a, false, b,
                                                 (short)0, c, false, false);
}

// 32-byte global -> LDS tile copy (two B128 transfers), async when available.
__device__ __forceinline__ void stage32(const bf16* g, bf16* s) {
#ifdef HAVE_ASYNC_LDS
  __builtin_amdgcn_global_load_async_to_lds_b128(
      (AS1 i32x4*)(g),     (AS3 i32x4*)(s),     0, 0);
  __builtin_amdgcn_global_load_async_to_lds_b128(
      (AS1 i32x4*)(g + 8), (AS3 i32x4*)(s + 8), 0, 0);
#else
  const f32x4* gv = reinterpret_cast<const f32x4*>(g);
  f32x4* sv = reinterpret_cast<f32x4*>(s);
  sv[0] = gv[0]; sv[1] = gv[1];
#endif
}

__device__ __forceinline__ void stage_wait() {
#ifdef HAVE_ASYNC_LDS
  __builtin_amdgcn_s_wait_asynccnt(0);
#endif
}

// ---------------------------------------------------------------------------
// f32 -> bf16 conversion (grid-stride)
// ---------------------------------------------------------------------------
__global__ void cvt_bf16_kernel(const float* __restrict__ in,
                                bf16* __restrict__ out, long n) {
  long i = (long)blockIdx.x * blockDim.x + threadIdx.x;
  long stride = (long)gridDim.x * blockDim.x;
  for (; i < n; i += stride) out[i] = (bf16)in[i];
}

// ---------------------------------------------------------------------------
// GEMM: C[M,N] = A[M,K] * B[N,K]^T   (A, B row-major bf16, C f32)
// 256 threads = 8 waves. WG tile 128x128, wave tile 32x64, K-step 32.
// ---------------------------------------------------------------------------
__global__ __launch_bounds__(256) void gemm_bf16_nt(
    const bf16* __restrict__ A, const bf16* __restrict__ Bm,
    float* __restrict__ Cm, int M, int N, int K) {
  __shared__ __attribute__((aligned(16))) bf16 As[128][40];
  __shared__ __attribute__((aligned(16))) bf16 Bs[128][40];

  const int tid  = threadIdx.x;
  const int lane = tid & 31;
  const int wave = tid >> 5;
  const int half = lane >> 4;
  const int l15  = lane & 15;

  const int bm = blockIdx.x * 128;
  const int bn = blockIdx.y * 128;
  const int mw = (wave >> 1) * 32;
  const int nw = (wave & 1) * 64;

  const int lr = tid >> 1;          // 0..127 : tile row loaded by this thread
  const int lc = (tid & 1) * 16;    // 0 or 16

  f32x8 acc[2][4] = {};

  for (int kb = 0; kb < K; kb += 32) {
    __syncthreads();
    {
      stage32(A  + (long)(bm + lr) * K + kb + lc, &As[lr][lc]);
      stage32(Bm + (long)(bn + lr) * K + kb + lc, &Bs[lr][lc]);
      if (kb + 32 < K) {   // gfx1250 global_prefetch of next K tile
        __builtin_prefetch(A + (long)(bm + lr) * K + kb + 32 + lc);
        __builtin_prefetch(Bm + (long)(bn + lr) * K + kb + 32 + lc);
      }
      stage_wait();
    }
    __syncthreads();

    bf16x16 af[2], bfr[4];
#pragma unroll
    for (int i = 0; i < 2; ++i) {   // A frag: per-lane chunks at k=8*half, 16+8*half
      FragU f;
      f.f[0] = *reinterpret_cast<const f32x4*>(&As[mw + i * 16 + l15][8 * half]);
      f.f[1] = *reinterpret_cast<const f32x4*>(&As[mw + i * 16 + l15][16 + 8 * half]);
      af[i] = f.v;
    }
#pragma unroll
    for (int j = 0; j < 4; ++j) {   // B frag: per-lane chunks at k=8*half, 16+8*half
      FragU f;
      f.f[0] = *reinterpret_cast<const f32x4*>(&Bs[nw + j * 16 + l15][8 * half]);
      f.f[1] = *reinterpret_cast<const f32x4*>(&Bs[nw + j * 16 + l15][16 + 8 * half]);
      bfr[j] = f.v;
    }
#pragma unroll
    for (int i = 0; i < 2; ++i)
#pragma unroll
      for (int j = 0; j < 4; ++j)
        acc[i][j] = wmma_bf16(af[i], bfr[j], acc[i][j]);
  }

#pragma unroll
  for (int i = 0; i < 2; ++i)
#pragma unroll
    for (int j = 0; j < 4; ++j)
#pragma unroll
      for (int r = 0; r < 8; ++r) {
        int m = bm + mw + i * 16 + r + 8 * half;   // C layout: VGPR r -> row r (+8 upper half)
        int n = bn + nw + j * 16 + l15;            // column = lane&15
        Cm[(long)m * N + n] = acc[i][j][r];
      }
}

// ---------------------------------------------------------------------------
// RoPE on q: qraw (B*T, H*64) f32 -> qbf (B,H,T,64) bf16
// ---------------------------------------------------------------------------
__global__ void rope_q_kernel(const float* __restrict__ qraw,
                              const float* __restrict__ cosT,
                              const float* __restrict__ sinT,
                              bf16* __restrict__ qbf, long n) {
  long i = (long)blockIdx.x * blockDim.x + threadIdx.x;
  if (i >= n) return;
  int p = i & 31; long t1 = i >> 5;
  int h = (int)(t1 % NH); t1 /= NH;
  int t = (int)(t1 % SEQ); int b = (int)(t1 / SEQ);
  const float* src = qraw + ((long)(b * SEQ + t) * NH + h) * HD;
  float x1 = src[2 * p], x2 = src[2 * p + 1];
  float c = cosT[t * HD + 2 * p], s = sinT[t * HD + 2 * p];
  float o1 = x1 * c - x2 * s;
  float o2 = x2 * c + x1 * s;
  bf16* dst = qbf + ((long)(b * NH + h) * SEQ + t) * HD;
  dst[2 * p] = (bf16)o1; dst[2 * p + 1] = (bf16)o2;
}

// RoPE on k: also emits the f32 (B,Hk,T,64) output tensor (return value 2)
__global__ void rope_k_kernel(const float* __restrict__ kraw,
                              const float* __restrict__ cosT,
                              const float* __restrict__ sinT,
                              bf16* __restrict__ kbf, float* __restrict__ kout, long n) {
  long i = (long)blockIdx.x * blockDim.x + threadIdx.x;
  if (i >= n) return;
  int p = i & 31; long t1 = i >> 5;
  int h = (int)(t1 % NKV); t1 /= NKV;
  int t = (int)(t1 % SEQ); int b = (int)(t1 / SEQ);
  const float* src = kraw + ((long)(b * SEQ + t) * NKV + h) * HD;
  float x1 = src[2 * p], x2 = src[2 * p + 1];
  float c = cosT[t * HD + 2 * p], s = sinT[t * HD + 2 * p];
  float o1 = x1 * c - x2 * s;
  float o2 = x2 * c + x1 * s;
  long di = ((long)(b * NKV + h) * SEQ + t) * HD;
  kbf[di + 2 * p] = (bf16)o1; kbf[di + 2 * p + 1] = (bf16)o2;
  kout[di + 2 * p] = o1;      kout[di + 2 * p + 1] = o2;
}

// v: repack (B*T, Hk*64) -> (B,Hk,T,64), bf16 for attention + f32 output (value 3)
__global__ void pack_v_kernel(const float* __restrict__ vraw,
                              bf16* __restrict__ vbf, float* __restrict__ vout, long n) {
  long i = (long)blockIdx.x * blockDim.x + threadIdx.x;
  if (i >= n) return;
  int d = i & 63; long t1 = i >> 6;
  int h = (int)(t1 % NKV); t1 /= NKV;
  int t = (int)(t1 % SEQ); int b = (int)(t1 / SEQ);
  float v = vraw[((long)(b * SEQ + t) * NKV + h) * HD + d];
  long di = ((long)(b * NKV + h) * SEQ + t) * HD + d;
  vbf[di] = (bf16)v; vout[di] = v;
}

// ---------------------------------------------------------------------------
// Flash attention: 128 threads (4 waves). 64 q rows / WG, 16 rows / wave.
// Online softmax over 32-key blocks; QK^T and PV on WMMA.
// ---------------------------------------------------------------------------
__global__ __launch_bounds__(128) void attn_fwd_kernel(
    const bf16* __restrict__ Qm,   // (B*H, T, 64)
    const bf16* __restrict__ Km,   // (B*Hk, T, 64)
    const bf16* __restrict__ Vm,   // (B*Hk, T, 64)
    bf16* __restrict__ Om)         // (B*T, H*64)
{
  __shared__ __attribute__((aligned(16))) bf16 Ks[32][72];      // keys x d
  __shared__ __attribute__((aligned(16))) bf16 Vt[64][40];      // d x keys (transposed)
  __shared__ __attribute__((aligned(16))) bf16 Ps[4][16][40];   // per-wave P staging

  const int tid  = threadIdx.x;
  const int lane = tid & 31;
  const int wave = tid >> 5;
  const int half = lane >> 4;
  const int l15  = lane & 15;

  const int bh    = blockIdx.x;          // b*H + h
  const int b     = bh / NH;
  const int h     = bh % NH;
  const int kvh   = h / NREP;
  const int qtile = blockIdx.y;          // 64 rows per tile

  const bf16* qbase = Qm + ((long)bh * SEQ + qtile * 64) * HD;
  const bf16* kbase = Km + ((long)(b * NKV + kvh) * SEQ) * HD;
  const bf16* vbase = Vm + ((long)(b * NKV + kvh) * SEQ) * HD;

  // Q fragments (16 rows x 64 d = two 16x32 A fragments), straight from global
  bf16x16 qa[2];
  {
    const bf16* qr = qbase + (long)(wave * 16 + l15) * HD;
#pragma unroll
    for (int c = 0; c < 2; ++c) {
      FragU f;
      f.f[0] = *reinterpret_cast<const f32x4*>(qr + c * 32 + 8 * half);
      f.f[1] = *reinterpret_cast<const f32x4*>(qr + c * 32 + 16 + 8 * half);
      qa[c] = f.v;
    }
  }

  f32x8 oacc[4] = {};
  float mrow[8], lrow[8];
#pragma unroll
  for (int r = 0; r < 8; ++r) { mrow[r] = -__builtin_inff(); lrow[r] = 0.f; }

  const int q0  = qtile * 64 + wave * 16;  // this wave's first global q row
  const int nkb = 2 * qtile + 2;           // causal: key blocks needed by this WG

  for (int kb = 0; kb < nkb; ++kb) {
    __syncthreads();
    { // cooperative load: K tile via async-to-LDS, V tile transposed via VGPRs
      int r = tid >> 2;            // key 0..31
      int c = (tid & 3) * 16;      // d chunk
      stage32(kbase + (long)(kb * 32 + r) * HD + c, &Ks[r][c]);
      FragU fv;
      const f32x4* gv = reinterpret_cast<const f32x4*>(vbase + (long)(kb * 32 + r) * HD + c);
      fv.f[0] = gv[0]; fv.f[1] = gv[1];
#pragma unroll
      for (int e = 0; e < 16; ++e) Vt[c + e][r] = fv.v[e];
      stage_wait();
    }
    __syncthreads();

    // scores S = Q K^T for 32 keys (two 16-key sub-tiles), scaled + causal-masked
    f32x8 sc[2];
#pragma unroll
    for (int s = 0; s < 2; ++s) {
      f32x8 a = {};
#pragma unroll
      for (int c = 0; c < 2; ++c) {  // B frag of K^T: key=lane&15, d chunks at 8*half, 16+8*half
        FragU f;
        f.f[0] = *reinterpret_cast<const f32x4*>(&Ks[s * 16 + l15][c * 32 + 8 * half]);
        f.f[1] = *reinterpret_cast<const f32x4*>(&Ks[s * 16 + l15][c * 32 + 16 + 8 * half]);
        a = wmma_bf16(qa[c], f.v, a);
      }
      int key = kb * 32 + s * 16 + l15;
#pragma unroll
      for (int r = 0; r < 8; ++r) {
        int qr = q0 + r + 8 * half;
        float v = a[r] * 0.125f;              // 1/sqrt(64)
        a[r] = (key <= qr) ? v : -__builtin_inff();
      }
      sc[s] = a;
    }

    // online softmax: rows live in VGPR index r, columns across 16 lanes
#pragma unroll
    for (int r = 0; r < 8; ++r) {
      float v = fmaxf(sc[0][r], sc[1][r]);
#pragma unroll
      for (int off = 1; off < 16; off <<= 1) v = fmaxf(v, __shfl_xor(v, off, 32));
      float nm   = fmaxf(mrow[r], v);
      float corr = __expf(mrow[r] - nm);
      float p0   = __expf(sc[0][r] - nm);
      float p1   = __expf(sc[1][r] - nm);
      float sum  = p0 + p1;
#pragma unroll
      for (int off = 1; off < 16; off <<= 1) sum += __shfl_xor(sum, off, 32);
      lrow[r] = lrow[r] * corr + sum;
      mrow[r] = nm;
#pragma unroll
      for (int j = 0; j < 4; ++j) oacc[j][r] *= corr;
      Ps[wave][r + 8 * half][l15]      = (bf16)p0;   // C layout -> LDS (A layout staging)
      Ps[wave][r + 8 * half][16 + l15] = (bf16)p1;
    }
    __syncthreads();   // uniform across WG (nkb is block-uniform)

    // O += P(16x32) * V(32x64): A frag of P from LDS, B frags from transposed V
    FragU fp;
    fp.f[0] = *reinterpret_cast<const f32x4*>(&Ps[wave][l15][8 * half]);
    fp.f[1] = *reinterpret_cast<const f32x4*>(&Ps[wave][l15][16 + 8 * half]);
#pragma unroll
    for (int j = 0; j < 4; ++j) {
      FragU fb;
      fb.f[0] = *reinterpret_cast<const f32x4*>(&Vt[j * 16 + l15][8 * half]);
      fb.f[1] = *reinterpret_cast<const f32x4*>(&Vt[j * 16 + l15][16 + 8 * half]);
      oacc[j] = wmma_bf16(fp.v, fb.v, oacc[j]);
    }
  }

  // normalize and store attention output activation (B*T, H*64) bf16
#pragma unroll
  for (int r = 0; r < 8; ++r) {
    float inv = 1.f / lrow[r];
    long row = (long)b * SEQ + qtile * 64 + wave * 16 + r + 8 * half;
#pragma unroll
    for (int j = 0; j < 4; ++j)
      Om[row * (long)(NH * HD) + h * HD + j * 16 + l15] = (bf16)(oacc[j][r] * inv);
  }
}

// ---------------------------------------------------------------------------
// Host launcher
// ---------------------------------------------------------------------------
extern "C" void kernel_launch(void* const* d_in, const int* in_sizes, int n_in,
                              void* d_out, int out_size, void* d_ws, size_t ws_size,
                              hipStream_t stream) {
  (void)in_sizes; (void)n_in; (void)out_size; (void)ws_size;
  const float* x    = (const float*)d_in[0];
  const float* cosT = (const float*)d_in[1];
  const float* sinT = (const float*)d_in[2];
  const float* wq   = (const float*)d_in[3];
  const float* wk   = (const float*)d_in[4];
  const float* wv   = (const float*)d_in[5];
  const float* wo   = (const float*)d_in[6];
  float* out = (float*)d_out;

  const long MT = (long)BATCH * SEQ;  // 4096 rows
  char* ws = (char*)d_ws;
  auto carve = [&](size_t bytes) -> char* {
    char* p = ws; ws += (bytes + 255) & ~(size_t)255; return p;
  };
  bf16*  x_bf  = (bf16*)carve(MT * CDIM * 2);
  bf16*  wq_bf = (bf16*)carve((long)NH * HD * CDIM * 2);
  bf16*  wk_bf = (bf16*)carve((long)NKV * HD * CDIM * 2);
  bf16*  wv_bf = (bf16*)carve((long)NKV * HD * CDIM * 2);
  bf16*  wo_bf = (bf16*)carve((long)CDIM * NH * HD * 2);
  float* qraw  = (float*)carve(MT * NH * HD * 4);
  float* kraw  = (float*)carve(MT * NKV * HD * 4);
  float* vraw  = (float*)carve(MT * NKV * HD * 4);
  bf16*  q_bf  = (bf16*)carve(MT * NH * HD * 2);
  bf16*  k_bf  = (bf16*)carve(MT * NKV * HD * 2);
  bf16*  v_bf  = (bf16*)carve(MT * NKV * HD * 2);
  bf16*  o_bf  = (bf16*)carve(MT * NH * HD * 2);

  // 1) f32 -> bf16
  cvt_bf16_kernel<<<2048, 256, 0, stream>>>(x,  x_bf,  MT * CDIM);
  cvt_bf16_kernel<<<2048, 256, 0, stream>>>(wq, wq_bf, (long)NH * HD * CDIM);
  cvt_bf16_kernel<<<512,  256, 0, stream>>>(wk, wk_bf, (long)NKV * HD * CDIM);
  cvt_bf16_kernel<<<512,  256, 0, stream>>>(wv, wv_bf, (long)NKV * HD * CDIM);
  cvt_bf16_kernel<<<2048, 256, 0, stream>>>(wo, wo_bf, (long)CDIM * NH * HD);

  // 2) QKV projections (x @ w.T)
  gemm_bf16_nt<<<dim3(4096 / 128, 2048 / 128), 256, 0, stream>>>(x_bf, wq_bf, qraw, 4096, 2048, 2048);
  gemm_bf16_nt<<<dim3(4096 / 128,  256 / 128), 256, 0, stream>>>(x_bf, wk_bf, kraw, 4096,  256, 2048);
  gemm_bf16_nt<<<dim3(4096 / 128,  256 / 128), 256, 0, stream>>>(x_bf, wv_bf, vraw, 4096,  256, 2048);

  // 3) RoPE + repack (+ emit f32 k and v outputs)
  float* k_out = out + MT * CDIM;
  float* v_out = k_out + (long)BATCH * NKV * SEQ * HD;
  long nq = MT * NH * (HD / 2);
  rope_q_kernel<<<(unsigned)((nq + 255) / 256), 256, 0, stream>>>(qraw, cosT, sinT, q_bf, nq);
  long nk = MT * NKV * (HD / 2);
  rope_k_kernel<<<(unsigned)((nk + 255) / 256), 256, 0, stream>>>(kraw, cosT, sinT, k_bf, k_out, nk);
  long nv = MT * NKV * HD;
  pack_v_kernel<<<(unsigned)((nv + 255) / 256), 256, 0, stream>>>(vraw, v_bf, v_out, nv);

  // 4) causal flash attention
  attn_fwd_kernel<<<dim3(BATCH * NH, SEQ / 64), 128, 0, stream>>>(q_bf, k_bf, v_bf, o_bf);

  // 5) output projection (o @ wo.T) -> out
  gemm_bf16_nt<<<dim3(4096 / 128, 2048 / 128), 256, 0, stream>>>(o_bf, wo_bf, out, 4096, 2048, 2048);
}
